// Attention_70111046140012
// MI455X (gfx1250) — matrix operand order repaired and checked by reading, not compiled, not run on hardware
//
#include <hip/hip_runtime.h>

// ---------------------------------------------------------------------------
// Attention block for gfx1250 (MI455X): QKV GEMM + rope + flash-attn + proj.
// GEMMs: v_wmma_f32_16x16x32_f16 (wave32), software-pipelined K loop.
// Flash attention: K/V tiles double-buffered in LDS via
// global_load_async_to_lds_b128 (ASYNCcnt) overlapped with WMMA compute.
// B=2, N=2048, D=768, H=12, HD=64
// ---------------------------------------------------------------------------

typedef __attribute__((ext_vector_type(16))) _Float16 v16h;
typedef __attribute__((ext_vector_type(8)))  _Float16 v8h;
typedef __attribute__((ext_vector_type(8)))  float    v8f;

static constexpr int Bb = 2, Nn = 2048, Dd = 768, Hh = 12, HD = 64;

#define WMMA16x16x32(a, b, c) \
    __builtin_amdgcn_wmma_f32_16x16x32_f16(false, (a), false, (b), (short)0, (c), false, false)

// ---- fragment loaders (contiguous k = 16h + i, same order for A and B, wave32) ---------------
// A 16x32 (MxK), row-major, ld = K stride in elements.
// lane: row = lane&15 ; halves: K = i + (lane<16 ? 0 : 16)
__device__ __forceinline__ v16h load_a_frag(const _Float16* base, int ld) {
    const int lane = (int)(threadIdx.x & 31u);
    const _Float16* p = base + (size_t)(lane & 15) * ld + ((lane & 16) ? 16 : 0);
    v8h lo = *reinterpret_cast<const v8h*>(p);
    v8h hi = *reinterpret_cast<const v8h*>(p + 8);
    v16h r;
#pragma unroll
    for (int i = 0; i < 8; ++i) { r[i] = lo[i]; r[i + 8] = hi[i]; }
    return r;
}

// B 32x16 (KxN) sourced from an N-major (row = n, contiguous K) array, ld = K stride.
// lane: col = lane&15 ; halves: K = i + (lane<16 ? 0 : 16)
__device__ __forceinline__ v16h load_b_frag(const _Float16* base, int ld) {
    const int lane = (int)(threadIdx.x & 31u);
    const _Float16* p = base + (size_t)(lane & 15) * ld + ((lane & 16) ? 16 : 0);
    v8h lo = *reinterpret_cast<const v8h*>(p);
    v8h hi = *reinterpret_cast<const v8h*>(p + 8);
    v16h r;
#pragma unroll
    for (int i = 0; i < 8; ++i) { r[i] = lo[i]; r[i + 8] = hi[i]; }
    return r;
}

// ---- kernel 1: fp32 -> f16 --------------------------------------------------
__global__ void f32_to_f16(const float* __restrict__ in, _Float16* __restrict__ out, int n) {
    int i = blockIdx.x * blockDim.x + threadIdx.x;
    if (i < n) out[i] = (_Float16)in[i];
}

// ---- kernel 2/5: C(MxN) = A(MxK) @ W(NxK)^T (+bias), f32 out ---------------
// 128 threads = 4 waves; wave w -> rows [by*128 + 32w, +32), cols [bx*64, +64)
// K loop software-pipelined (next fragments loaded before current WMMAs).
__global__ __launch_bounds__(128) void gemm_xwt_wmma(
    const _Float16* __restrict__ A, const _Float16* __restrict__ W,
    const float* __restrict__ bias, float* __restrict__ C,
    int M, int N, int K)
{
    const int t    = (int)threadIdx.x;
    const int wave = t >> 5;
    const int lane = t & 31;
    const int m0 = blockIdx.y * 128 + wave * 32;
    const int n0 = blockIdx.x * 64;

    const _Float16* Ab0 = A + (size_t)m0 * K;
    const _Float16* Ab1 = A + (size_t)(m0 + 16) * K;
    const _Float16* Wb0 = W + (size_t)(n0 +  0) * K;
    const _Float16* Wb1 = W + (size_t)(n0 + 16) * K;
    const _Float16* Wb2 = W + (size_t)(n0 + 32) * K;
    const _Float16* Wb3 = W + (size_t)(n0 + 48) * K;

    v8f acc[2][4] = {};
    v16h a0 = load_a_frag(Ab0, K);
    v16h a1 = load_a_frag(Ab1, K);
    v16h b0 = load_b_frag(Wb0, K);
    v16h b1 = load_b_frag(Wb1, K);
    v16h b2 = load_b_frag(Wb2, K);
    v16h b3 = load_b_frag(Wb3, K);

    for (int k0 = 0; k0 + 32 < K; k0 += 32) {
        const int kn = k0 + 32;
        if (kn + 32 < K) {  // prefetch tile-after-next toward L2
            __builtin_prefetch(A + (size_t)(blockIdx.y * 128 + t) * K + kn + 32, 0, 1);
            if (t < 64)
                __builtin_prefetch(W + (size_t)(n0 + t) * K + kn + 32, 0, 1);
        }
        v16h na0 = load_a_frag(Ab0 + kn, K);
        v16h na1 = load_a_frag(Ab1 + kn, K);
        v16h nb0 = load_b_frag(Wb0 + kn, K);
        v16h nb1 = load_b_frag(Wb1 + kn, K);
        v16h nb2 = load_b_frag(Wb2 + kn, K);
        v16h nb3 = load_b_frag(Wb3 + kn, K);

        acc[0][0] = WMMA16x16x32(a0, b0, acc[0][0]);
        acc[1][0] = WMMA16x16x32(a1, b0, acc[1][0]);
        acc[0][1] = WMMA16x16x32(a0, b1, acc[0][1]);
        acc[1][1] = WMMA16x16x32(a1, b1, acc[1][1]);
        acc[0][2] = WMMA16x16x32(a0, b2, acc[0][2]);
        acc[1][2] = WMMA16x16x32(a1, b2, acc[1][2]);
        acc[0][3] = WMMA16x16x32(a0, b3, acc[0][3]);
        acc[1][3] = WMMA16x16x32(a1, b3, acc[1][3]);

        a0 = na0; a1 = na1; b0 = nb0; b1 = nb1; b2 = nb2; b3 = nb3;
    }
    acc[0][0] = WMMA16x16x32(a0, b0, acc[0][0]);
    acc[1][0] = WMMA16x16x32(a1, b0, acc[1][0]);
    acc[0][1] = WMMA16x16x32(a0, b1, acc[0][1]);
    acc[1][1] = WMMA16x16x32(a1, b1, acc[1][1]);
    acc[0][2] = WMMA16x16x32(a0, b2, acc[0][2]);
    acc[1][2] = WMMA16x16x32(a1, b2, acc[1][2]);
    acc[0][3] = WMMA16x16x32(a0, b3, acc[0][3]);
    acc[1][3] = WMMA16x16x32(a1, b3, acc[1][3]);

    const int cl = lane & 15;
    const int rb = (lane & 16) ? 8 : 0;
#pragma unroll
    for (int s = 0; s < 2; ++s) {
#pragma unroll
        for (int f = 0; f < 4; ++f) {
            const int col = n0 + f * 16 + cl;
            const float bz = bias ? bias[col] : 0.0f;
#pragma unroll
            for (int j = 0; j < 8; ++j)
                C[(size_t)(m0 + s * 16 + rb + j) * N + col] = acc[s][f][j] + bz;
        }
    }
}

// ---- kernel 3: rotary + scale + scatter to q/k/vt (f16) --------------------
__global__ void rope_scatter(const float* __restrict__ qkv,   // [B*N][3D]
                             const float* __restrict__ pos,   // [N][HD]
                             _Float16* __restrict__ qh,       // [BH][N][HD]
                             _Float16* __restrict__ kh,       // [BH][N][HD]
                             _Float16* __restrict__ vth)      // [BH][HD][N]
{
    int idx = blockIdx.x * blockDim.x + threadIdx.x;
    int i  = idx & 31;
    int n  = (idx >> 5) & (Nn - 1);
    int bh = idx >> 16;                 // 5 + 11 bits
    if (bh >= Bb * Hh) return;
    int h = bh % Hh, b = bh / Hh;

    size_t base = (size_t)(b * Nn + n) * (3 * Dd) + h * HD;
    float q1 = qkv[base + i],          q2 = qkv[base + i + 32];
    float k1 = qkv[base + Dd + i],     k2 = qkv[base + Dd + i + 32];
    float v1 = qkv[base + 2 * Dd + i], v2 = qkv[base + 2 * Dd + i + 32];
    float t1 = pos[n * HD + i],        t2 = pos[n * HD + i + 32];
    float c1 = __cosf(t1), s1 = __sinf(t1);
    float c2 = __cosf(t2), s2 = __sinf(t2);
    const float sc = 0.125f;            // 1/sqrt(HD), folded into q
    float qo1 = (q1 * c1 - q2 * s1) * sc;
    float qo2 = (q2 * c2 + q1 * s2) * sc;
    float ko1 = k1 * c1 - k2 * s1;
    float ko2 = k2 * c2 + k1 * s2;

    size_t qk = ((size_t)bh * Nn + n) * HD;
    qh[qk + i]      = (_Float16)qo1;
    qh[qk + i + 32] = (_Float16)qo2;
    kh[qk + i]      = (_Float16)ko1;
    kh[qk + i + 32] = (_Float16)ko2;
    size_t vb = (size_t)bh * HD * Nn;
    vth[vb + (size_t)i * Nn + n]        = (_Float16)v1;
    vth[vb + (size_t)(i + 32) * Nn + n] = (_Float16)v2;
}

// ---- kernel 4: flash attention (double-buffered async LDS + WMMA) ----------
// block = 4 waves; wave -> 16 query rows; block shares K/V tiles staged in LDS
// by the async copy engine; tile i+1 DMA overlaps tile i compute.
__global__ __launch_bounds__(128) void flash_attn_wmma(
    const _Float16* __restrict__ qh, const _Float16* __restrict__ kh,
    const _Float16* __restrict__ vth, _Float16* __restrict__ ao)  // ao: [B][N][D]
{
    __shared__ _Float16 sK[2][32 * 64];    // [key][hd]   2 x 4KB
    __shared__ _Float16 sV[2][64 * 32];    // [hd][key]   2 x 4KB
    __shared__ _Float16 sP[4 * 16 * 32];   // per-wave P  4KB
    const int t    = (int)threadIdx.x;
    const int wave = t >> 5;
    const int lane = t & 31;
    const int bh = blockIdx.y;
    const int b = bh / Hh, h = bh % Hh;
    const int q0 = blockIdx.x * 64 + wave * 16;

    const _Float16* qb = qh  + (size_t)bh * Nn * HD;
    const _Float16* kb = kh  + (size_t)bh * Nn * HD;
    const _Float16* vb = vth + (size_t)bh * HD * Nn;
    _Float16* lp = sP + wave * (16 * 32);

    const v16h aq0 = load_a_frag(qb + (size_t)q0 * HD + 0, HD);
    const v16h aq1 = load_a_frag(qb + (size_t)q0 * HD + 32, HD);

    float mrun[8], lrun[8];
#pragma unroll
    for (int j = 0; j < 8; ++j) { mrun[j] = -3.0e38f; lrun[j] = 0.0f; }
    v8f acc[4] = {};

    const int cl = lane & 15;
    const int rb = (lane & 16) ? 8 : 0;

    // low 32 bits of a flat shared-memory address = workgroup LDS byte offset
    const uint32_t ldsK[2] = { (uint32_t)(uintptr_t)&sK[0][0], (uint32_t)(uintptr_t)&sK[1][0] };
    const uint32_t ldsV[2] = { (uint32_t)(uintptr_t)&sV[0][0], (uint32_t)(uintptr_t)&sV[1][0] };

    // async stage of one 32-key tile: K 32x64 (4KB) + V^T 64x32 (4KB);
    // 256 chunks of 16B each per matrix; 128 threads x 2 chunks.
    auto stage_tile = [&](int k0, int buf) {
#pragma unroll
        for (int rep = 0; rep < 2; ++rep) {
            const int idx = t + rep * 128;                       // 0..255
            const _Float16* gk = kb + (size_t)k0 * HD + (size_t)idx * 8;
            const uint32_t   dk = ldsK[buf] + (uint32_t)idx * 16u;
            asm volatile("global_load_async_to_lds_b128 %0, %1, off"
                         :: "v"(dk), "v"(gk) : "memory");
            const _Float16* gv = vb + (size_t)(idx >> 2) * Nn + k0 + (idx & 3) * 8;
            const uint32_t   dv = ldsV[buf] + (uint32_t)idx * 16u;
            asm volatile("global_load_async_to_lds_b128 %0, %1, off"
                         :: "v"(dv), "v"(gv) : "memory");
        }
    };

    int cur = 0;
    stage_tile(0, 0);
    for (int k0 = 0; k0 < Nn; k0 += 32) {
        asm volatile("s_wait_asynccnt 0" ::: "memory");  // tile k0 resident
        __syncthreads();                                 // ..across all waves
        if (k0 + 32 < Nn) {
            stage_tile(k0 + 32, cur ^ 1);                // DMA next tile now,
            __builtin_prefetch(kb + (size_t)(k0 + 64) * HD + (size_t)t * 16, 0, 1);
        }                                                // overlap with WMMAs

        const _Float16* cK = &sK[cur][0];
        const _Float16* cV = &sV[cur][0];

        // ---- S(16x32) = Q(16x64) @ K^T(64x32), fragments from LDS ----------
        v8f s0 = {}, s1 = {};
        {
            v16h bk0 = load_b_frag(cK, 64);
            s0 = WMMA16x16x32(aq0, bk0, s0);
            v16h bk1 = load_b_frag(cK + 32, 64);
            s0 = WMMA16x16x32(aq1, bk1, s0);
            v16h bk2 = load_b_frag(cK + 16 * 64, 64);
            s1 = WMMA16x16x32(aq0, bk2, s1);
            v16h bk3 = load_b_frag(cK + 16 * 64 + 32, 64);
            s1 = WMMA16x16x32(aq1, bk3, s1);
        }
        // ---- online softmax per row (rows live across 16-lane groups) ------
#pragma unroll
        for (int j = 0; j < 8; ++j) {
            float tm = fmaxf(s0[j], s1[j]);
#pragma unroll
            for (int mk = 1; mk < 16; mk <<= 1)
                tm = fmaxf(tm, __shfl_xor(tm, mk, 32));
            float Mj = fmaxf(mrun[j], tm);
            float cf = __expf(mrun[j] - Mj);
            mrun[j] = Mj;
            float p0 = __expf(s0[j] - Mj);
            float p1 = __expf(s1[j] - Mj);
            float rs = p0 + p1;
#pragma unroll
            for (int mk = 1; mk < 16; mk <<= 1)
                rs += __shfl_xor(rs, mk, 32);
            lrun[j] = lrun[j] * cf + rs;
#pragma unroll
            for (int f = 0; f < 4; ++f) acc[f][j] *= cf;
            const int r = j + rb;
            lp[r * 32 + cl]      = (_Float16)p0;
            lp[r * 32 + cl + 16] = (_Float16)p1;
        }
        asm volatile("s_wait_dscnt 0" ::: "memory");   // P visible within wave
        // ---- O(16x64) += P(16x32) @ V(32x64), V from LDS -------------------
        v16h ap = load_a_frag(lp, 32);
#pragma unroll
        for (int f = 0; f < 4; ++f) {
            v16h bv = load_b_frag(cV + (size_t)(f * 16) * 32, 32);
            acc[f] = WMMA16x16x32(ap, bv, acc[f]);
        }
        cur ^= 1;
        // note: WMMA operand waits force all LDS reads of this buffer to have
        // completed before this wave reaches the next barrier -> safe reuse.
    }
    // ---- epilogue: normalize and scatter into [B][N][D] --------------------
#pragma unroll
    for (int j = 0; j < 8; ++j) {
        const float inv = 1.0f / lrun[j];
        const int nrow = q0 + rb + j;
#pragma unroll
        for (int f = 0; f < 4; ++f)
            ao[((size_t)b * Nn + nrow) * Dd + h * HD + f * 16 + cl] =
                (_Float16)(acc[f][j] * inv);
    }
}

// ---------------------------------------------------------------------------
extern "C" void kernel_launch(void* const* d_in, const int* in_sizes, int n_in,
                              void* d_out, int out_size, void* d_ws, size_t ws_size,
                              hipStream_t stream)
{
    const float* x     = (const float*)d_in[0];   // (B,N,D)
    const float* pos   = (const float*)d_in[1];   // (N,HD)
    const float* wqkv  = (const float*)d_in[2];   // (3D,D)
    const float* wproj = (const float*)d_in[3];   // (D,D)
    const float* bproj = (const float*)d_in[4];   // (D,)
    float* out = (float*)d_out;                   // (B,N,D)

    const int M = Bb * Nn;                        // 4096
    char* ws = (char*)d_ws;
    size_t off = 0;
    auto alloc = [&](size_t bytes) -> void* {
        void* p = ws + off;
        off = (off + bytes + 255) & ~(size_t)255;
        return p;
    };
    _Float16* xh  = (_Float16*)alloc((size_t)M * Dd * 2);
    _Float16* wqh = (_Float16*)alloc((size_t)3 * Dd * Dd * 2);
    _Float16* wph = (_Float16*)alloc((size_t)Dd * Dd * 2);
    float*    qkv = (float*)   alloc((size_t)M * 3 * Dd * 4);
    _Float16* qhb = (_Float16*)alloc((size_t)M * Dd * 2);
    _Float16* khb = (_Float16*)alloc((size_t)M * Dd * 2);
    _Float16* vtb = (_Float16*)alloc((size_t)M * Dd * 2);
    _Float16* aoh = (_Float16*)alloc((size_t)M * Dd * 2);

    // 1) precision convert
    int n1 = M * Dd;
    f32_to_f16<<<(n1 + 255) / 256, 256, 0, stream>>>(x, xh, n1);
    int n2 = 3 * Dd * Dd;
    f32_to_f16<<<(n2 + 255) / 256, 256, 0, stream>>>(wqkv, wqh, n2);
    int n3 = Dd * Dd;
    f32_to_f16<<<(n3 + 255) / 256, 256, 0, stream>>>(wproj, wph, n3);

    // 2) QKV GEMM: (4096x768) @ (768x2304) -> f32
    dim3 g1(3 * Dd / 64, M / 128);                // 36 x 32
    gemm_xwt_wmma<<<g1, 128, 0, stream>>>(xh, wqh, nullptr, qkv, M, 3 * Dd, Dd);

    // 3) rotary + scatter
    int nr = Bb * Hh * Nn * 32;
    rope_scatter<<<(nr + 255) / 256, 256, 0, stream>>>(qkv, pos, qhb, khb, vtb);

    // 4) flash attention
    dim3 g2(Nn / 64, Bb * Hh);                    // 32 x 24
    flash_attn_wmma<<<g2, 128, 0, stream>>>(qhb, khb, vtb, aoh);

    // 5) output projection with bias
    dim3 g3(Dd / 64, M / 128);                    // 12 x 32
    gemm_xwt_wmma<<<g3, 128, 0, stream>>>(aoh, wph, bproj, out, M, Dd, Dd);

    (void)in_sizes; (void)n_in; (void)out_size; (void)ws_size;
}
